// TransformerBlock_70738111365376
// MI455X (gfx1250) — matrix operand order repaired and checked by reading, not compiled, not run on hardware
//
#include <hip/hip_runtime.h>

typedef __bf16 v16bf __attribute__((ext_vector_type(16)));
typedef float  v8f   __attribute__((ext_vector_type(8)));
typedef unsigned int u32x4 __attribute__((ext_vector_type(4)));
typedef unsigned int u32x8 __attribute__((ext_vector_type(8)));

union Frag16 { v16bf v; unsigned int u[8]; };

#define DIMD 768
#define SEQ  1024
#define NB   8
#define NH   12
#define HD   64
#define D3   2304
#define HID  3072
#define NTOK (NB * SEQ)   // 8192

__device__ __forceinline__ unsigned short f32_to_bf16(float f) {
  unsigned int u = __float_as_uint(f);
  u += 0x7FFFu + ((u >> 16) & 1u);            // round-to-nearest-even
  return (unsigned short)(u >> 16);
}

__device__ __forceinline__ float gelu_tanh(float x) {
  float t = 0.7978845608028654f * (x + 0.044715f * x * x * x);
  return 0.5f * x * (1.0f + tanhf(t));
}

// ---- CDNA5 async global->LDS copy (ASYNCcnt-tracked) -----------------------
__device__ __forceinline__ void g2l_async_b128(unsigned lds_off, const void* gptr) {
  asm volatile("global_load_async_to_lds_b128 %0, %1, off"
               :: "v"(lds_off), "v"(gptr) : "memory");
}
__device__ __forceinline__ void wait_async0() {
  asm volatile("s_wait_asynccnt 0x0" ::: "memory");
}

// ---- CDNA5 Tensor Data Mover: 2D tile global->LDS (TENSORcnt-tracked) ------
// D# group 0: [1:0]=count(1), [63:32]=lds_addr, [120:64]=global_addr, [127:126]=type(2)
__device__ __forceinline__ u32x4 tdm_g0(unsigned lds_off, const void* gaddr) {
  unsigned long long ga = (unsigned long long)gaddr;
  u32x4 g;
  g.x = 1u;
  g.y = lds_off;
  g.z = (unsigned)ga;
  g.w = (unsigned)(ga >> 32) | (2u << 30);
  return g;
}
// D# group 1: ctrl (mask/data_size/pad), tensor_dim0/1, tile_dim0/1, dim0 stride
__device__ __forceinline__ u32x8 tdm_g1(unsigned ctrl, unsigned td0, unsigned td1,
                                        unsigned tile0, unsigned tile1,
                                        unsigned long long stride0) {
  u32x8 g;
  g[0] = ctrl;                                   // [17:16]=data_size, [20]=pad_en, ...
  g[1] = (td0 & 0xFFFFu) << 16;                  // [63:48] = tensor_dim0[15:0]
  g[2] = (td0 >> 16) | ((td1 & 0xFFFFu) << 16);  // dim0 hi | dim1 lo
  g[3] = (td1 >> 16) | (tile0 << 16);            // dim1 hi | tile_dim0
  g[4] = tile1 & 0xFFFFu;                        // tile_dim1 (tile_dim2 = 0)
  g[5] = (unsigned)stride0;                      // tensor_dim0_stride[31:0]
  g[6] = (unsigned)(stride0 >> 32) & 0xFFFFu;    // stride[47:32] (dim1 stride = 0)
  g[7] = 0u;
  return g;
}
__device__ __forceinline__ void tdm_load_2d(u32x4 g0, u32x8 g1) {
  asm volatile("tensor_load_to_lds %0, %1" :: "s"(g0), "s"(g1) : "memory");
}

// ---------------------------------------------------------------- f32 -> bf16
__global__ __launch_bounds__(256)
void k_cvt_bf16(const float* __restrict__ src, unsigned short* __restrict__ dst, int n) {
  int i = blockIdx.x * 256 + threadIdx.x;
  if (i < n) dst[i] = f32_to_bf16(src[i]);
}

// ---------------------------------------------------------------- LayerNorm
__global__ __launch_bounds__(256)
void k_layernorm(const float* __restrict__ x, const float* __restrict__ g,
                 const float* __restrict__ b, unsigned short* __restrict__ y, int rows) {
  int row  = blockIdx.x * 8 + (threadIdx.x >> 5);
  int lane = threadIdx.x & 31;
  if (row >= rows) return;
  const float* xr = x + (size_t)row * DIMD;
  float s = 0.f, s2 = 0.f;
  for (int i = lane; i < DIMD; i += 32) { float v = xr[i]; s += v; s2 += v * v; }
  #pragma unroll
  for (int m = 16; m > 0; m >>= 1) { s += __shfl_xor(s, m, 32); s2 += __shfl_xor(s2, m, 32); }
  float mu   = s  * (1.0f / DIMD);
  float var  = s2 * (1.0f / DIMD) - mu * mu;
  float rstd = rsqrtf(var + 1e-6f);
  unsigned short* yr = y + (size_t)row * DIMD;
  for (int i = lane; i < DIMD; i += 32)
    yr[i] = f32_to_bf16((xr[i] - mu) * rstd * g[i] + b[i]);
}

// ---------------------------------------------------------------- WMMA GEMM
// C[M,N] = A[M,K](bf16) * B[K,N](bf16) + bias, optional residual/GELU.
// Block tile 128x128, BK=32, 8 waves -> 64x32 each.
// A tile: per-lane async DMA into padded LDS; B tile: TDM descriptor DMA.
__global__ __launch_bounds__(256)
void k_gemm(const unsigned short* __restrict__ A, const unsigned short* __restrict__ B,
            const float* __restrict__ bias, const float* __restrict__ residual,
            float* __restrict__ outf, unsigned short* __restrict__ outb,
            int M, int N, int K, int act_gelu) {
  __shared__ __align__(16) unsigned short lsA[128 * 40];   // padded [m][k]
  __shared__ __align__(16) unsigned short lsBr[32 * 128];  // row-major [k][n]

  const int tid  = threadIdx.x;
  const int lane = tid & 31, half = lane >> 4, l16 = lane & 15;
  const int wave = tid >> 5, wm = wave >> 2, wn = wave & 3;
  const int n0 = blockIdx.x * 128, m0 = blockIdx.y * 128;

  v8f c[4][2];
  #pragma unroll
  for (int mi = 0; mi < 4; ++mi)
    #pragma unroll
    for (int ni = 0; ni < 2; ++ni)
      #pragma unroll
      for (int i = 0; i < 8; ++i) c[mi][ni][i] = 0.0f;

  const int am = tid >> 1, akc = (tid & 1) * 16;     // A staging coords
  const int bk = tid >> 3, bnc = (tid & 7) * 16;     // B prefetch coords
  const unsigned ldsa  = (unsigned)(uintptr_t)(lsA + am * 40 + akc);
  const unsigned ldsbr = (unsigned)(uintptr_t)lsBr;

  for (int k0 = 0; k0 < K; k0 += 32) {
    { // A tile 128x32 via per-lane async DMA straight into LDS
      const unsigned short* g = A + (size_t)(m0 + am) * K + k0 + akc;
      g2l_async_b128(ldsa,      g);
      g2l_async_b128(ldsa + 16, g + 8);
    }
    if (wave == 0) { // B tile 32x128 via Tensor Data Mover (one issue per WG)
      u32x4 g0 = tdm_g0(ldsbr, B + (size_t)k0 * N + n0);
      u32x8 g1 = tdm_g1(1u << 16 /*data_size=2B*/, (unsigned)N, (unsigned)K,
                        128u, 32u, (unsigned long long)N);
      tdm_load_2d(g0, g1);
    }
    if (k0 + 32 < K) { // cache hints for the next tiles (global_prefetch_b8)
      __builtin_prefetch(A + (size_t)(m0 + am) * K + k0 + 32 + akc, 0, 1);
      __builtin_prefetch(B + (size_t)(k0 + 32 + bk) * N + n0 + bnc, 0, 1);
    }
    wait_async0();                                 // A tile landed in LDS
    if (wave == 0) __builtin_amdgcn_s_wait_tensorcnt((short)0);  // B tile landed
    __syncthreads();

    Frag16 a[4], bf[2];
    #pragma unroll
    for (int mi = 0; mi < 4; ++mi) {
      const unsigned int* pa = (const unsigned int*)lsA + (wm * 64 + mi * 16 + l16) * 20;
      #pragma unroll
      for (int r = 0; r < 8; ++r) {
        int off = 8 * half + r;
        a[mi].u[r] = pa[off];
      }
    }
    #pragma unroll
    for (int ni = 0; ni < 2; ++ni) {
      const unsigned short* pb = lsBr + wn * 32 + ni * 16 + l16;
      #pragma unroll
      for (int r = 0; r < 8; ++r) {
        unsigned lo = pb[(2 * r +     16 * half) * 128];
        unsigned hi = pb[(2 * r + 1 + 16 * half) * 128];
        bf[ni].u[r] = lo | (hi << 16);
      }
    }
    #pragma unroll
    for (int mi = 0; mi < 4; ++mi)
      #pragma unroll
      for (int ni = 0; ni < 2; ++ni)
        c[mi][ni] = __builtin_amdgcn_wmma_f32_16x16x32_bf16(
            false, a[mi].v, false, bf[ni].v, (short)0, c[mi][ni], false, false);
    __syncthreads();
  }

  #pragma unroll
  for (int mi = 0; mi < 4; ++mi)
    #pragma unroll
    for (int ni = 0; ni < 2; ++ni)
      #pragma unroll
      for (int r = 0; r < 8; ++r) {
        int m = m0 + wm * 64 + mi * 16 + r + 8 * half;
        int n = n0 + wn * 32 + ni * 16 + l16;
        float v = c[mi][ni][r] + bias[n];
        if (residual) v += residual[(size_t)m * N + n];
        if (act_gelu) v = gelu_tanh(v);
        if (outf) outf[(size_t)m * N + n] = v;
        if (outb) outb[(size_t)m * N + n] = f32_to_bf16(v);
      }
}

// ---------------------------------------------------------------- attention
// qkv: bf16 [NTOK, 2304] laid out as (3, head, hd) on the fast axis.
// 4 waves/WG, each wave owns 16 query rows; flash-style over 32-key blocks.
__global__ __launch_bounds__(128)
void k_attn(const unsigned short* __restrict__ qkv, unsigned short* __restrict__ o) {
  __shared__ __align__(16) unsigned short lsP[4][16 * 40]; // per-wave P tile

  const int wave = threadIdx.x >> 5;
  const int lane = threadIdx.x & 31, half = lane >> 4, l16 = lane & 15;

  int wg   = blockIdx.x;          // b*NH*16 + h*16 + qblk
  int qblk = wg & 15;
  int bh   = wg >> 4;
  int h    = bh % NH;
  int b    = bh / NH;
  int qbase = qblk * 64 + wave * 16;

  // Q A-fragments (d = 0..63 split into two k=32 frags)
  Frag16 aq[2];
  {
    const unsigned short* qp = qkv + (size_t)(b * SEQ + qbase + l16) * D3 + h * HD;
    #pragma unroll
    for (int f = 0; f < 2; ++f)
      #pragma unroll
      for (int r = 0; r < 8; ++r) {
        int k = 16 * half + 2 * r;
        aq[f].u[r] = *(const unsigned int*)(qp + 32 * f + k);
      }
  }

  float mrow[8], lrow[8];
  v8f oacc[4];
  #pragma unroll
  for (int r = 0; r < 8; ++r) { mrow[r] = -1e30f; lrow[r] = 0.f; }
  #pragma unroll
  for (int fd = 0; fd < 4; ++fd)
    #pragma unroll
    for (int i = 0; i < 8; ++i) oacc[fd][i] = 0.f;

  unsigned short* P = &lsP[wave][0];
  const float sc = 0.125f; // hd^-0.5

  for (int kb = 0; kb < 32; ++kb) {
    int kbase = kb * 32;

    // S = Q K^T  (two 16-column tiles)
    v8f s[2];
    #pragma unroll
    for (int jt = 0; jt < 2; ++jt) {
      const unsigned short* kp =
          qkv + (size_t)(b * SEQ + kbase + 16 * jt + l16) * D3 + DIMD + h * HD;
      Frag16 bkf[2];
      #pragma unroll
      for (int f = 0; f < 2; ++f)
        #pragma unroll
        for (int r = 0; r < 8; ++r)
          bkf[f].u[r] = *(const unsigned int*)(kp + 32 * f + 2 * r + 16 * half);
      v8f cz = {0.f, 0.f, 0.f, 0.f, 0.f, 0.f, 0.f, 0.f};
      cz = __builtin_amdgcn_wmma_f32_16x16x32_bf16(false, aq[0].v, false, bkf[0].v,
                                                   (short)0, cz, false, false);
      cz = __builtin_amdgcn_wmma_f32_16x16x32_bf16(false, aq[1].v, false, bkf[1].v,
                                                   (short)0, cz, false, false);
      s[jt] = cz;
    }

    // online softmax (row m = r + 8*half lives across the 16 lanes of each half)
    float nm[8], alpha[8];
    #pragma unroll
    for (int r = 0; r < 8; ++r) {
      float v = fmaxf(s[0][r], s[1][r]) * sc;
      #pragma unroll
      for (int mk = 1; mk < 16; mk <<= 1) v = fmaxf(v, __shfl_xor(v, mk, 32));
      nm[r]    = fmaxf(mrow[r], v);
      alpha[r] = __expf(mrow[r] - nm[r]);
      mrow[r]  = nm[r];
    }
    #pragma unroll
    for (int r = 0; r < 8; ++r) {
      float p0 = __expf(s[0][r] * sc - nm[r]);
      float p1 = __expf(s[1][r] * sc - nm[r]);
      float ps = p0 + p1;
      #pragma unroll
      for (int mk = 1; mk < 16; mk <<= 1) ps += __shfl_xor(ps, mk, 32);
      lrow[r] = lrow[r] * alpha[r] + ps;
      int m = r + 8 * half;
      P[m * 40 + l16]      = f32_to_bf16(p0);
      P[m * 40 + 16 + l16] = f32_to_bf16(p1);
    }
    __syncthreads();

    // rescale running O
    #pragma unroll
    for (int fd = 0; fd < 4; ++fd)
      #pragma unroll
      for (int r = 0; r < 8; ++r) oacc[fd][r] *= alpha[r];

    // P as A-fragment (16x32)
    Frag16 ap;
    #pragma unroll
    for (int r = 0; r < 8; ++r) {
      int k = 16 * half + 2 * r;
      ap.u[r] = *(const unsigned int*)(P + l16 * 40 + k);
    }
    __syncthreads();

    // O += P * V  (four 16-wide d tiles)
    #pragma unroll
    for (int fd = 0; fd < 4; ++fd) {
      const unsigned short* vp = qkv + 2 * DIMD + h * HD + 16 * fd + l16;
      Frag16 bv;
      #pragma unroll
      for (int r = 0; r < 8; ++r) {
        int krow = 2 * r + 16 * half;
        unsigned int lo = vp[(size_t)(b * SEQ + kbase + krow) * D3];
        unsigned int hi = vp[(size_t)(b * SEQ + kbase + krow + 1) * D3];
        bv.u[r] = lo | (hi << 16);
      }
      oacc[fd] = __builtin_amdgcn_wmma_f32_16x16x32_bf16(false, ap.v, false, bv.v,
                                                         (short)0, oacc[fd], false, false);
    }
  }

  // normalize and emit o[b, n, h*64 + d] as bf16
  #pragma unroll
  for (int fd = 0; fd < 4; ++fd)
    #pragma unroll
    for (int r = 0; r < 8; ++r) {
      int m = r + 8 * half;
      float v = oacc[fd][r] / lrow[r];
      o[(size_t)(b * SEQ + qbase + m) * DIMD + h * HD + 16 * fd + l16] = f32_to_bf16(v);
    }
}

// ---------------------------------------------------------------- launcher
extern "C" void kernel_launch(void* const* d_in, const int* in_sizes, int n_in,
                              void* d_out, int out_size, void* d_ws, size_t ws_size,
                              hipStream_t stream) {
  const float* x     = (const float*)d_in[0];
  const float* ln1_s = (const float*)d_in[1];
  const float* ln1_b = (const float*)d_in[2];
  const float* qkv_w = (const float*)d_in[3];
  const float* qkv_b = (const float*)d_in[4];
  const float* proj_w= (const float*)d_in[5];
  const float* proj_b= (const float*)d_in[6];
  const float* ln2_s = (const float*)d_in[7];
  const float* ln2_b = (const float*)d_in[8];
  const float* fc1_w = (const float*)d_in[9];
  const float* fc1_b = (const float*)d_in[10];
  const float* fc2_w = (const float*)d_in[11];
  const float* fc2_b = (const float*)d_in[12];
  float* out = (float*)d_out;

  // workspace carve-up (halfword units)
  unsigned short* ws = (unsigned short*)d_ws;
  unsigned short* w_qkv = ws;                                    // 768*2304
  unsigned short* w_prj = w_qkv + (size_t)DIMD * D3;             // 768*768
  unsigned short* w_fc1 = w_prj + (size_t)DIMD * DIMD;           // 768*3072
  unsigned short* w_fc2 = w_fc1 + (size_t)DIMD * HID;            // 3072*768
  unsigned short* ybf   = w_fc2 + (size_t)HID * DIMD;            // 8192*768 (LN out / attn out)
  unsigned short* qkvb  = ybf   + (size_t)NTOK * DIMD;           // 8192*2304
  unsigned short* hbf   = qkvb  + (size_t)NTOK * D3;             // 8192*3072

  auto cvt = [&](const float* s, unsigned short* d, int n) {
    k_cvt_bf16<<<(n + 255) / 256, 256, 0, stream>>>(s, d, n);
  };
  cvt(qkv_w, w_qkv, DIMD * D3);
  cvt(proj_w, w_prj, DIMD * DIMD);
  cvt(fc1_w, w_fc1, DIMD * HID);
  cvt(fc2_w, w_fc2, HID * DIMD);

  // LN1 -> ybf
  k_layernorm<<<NTOK / 8, 256, 0, stream>>>(x, ln1_s, ln1_b, ybf, NTOK);

  // QKV: [8192,768] x [768,2304] -> qkvb (bf16)
  k_gemm<<<dim3(D3 / 128, NTOK / 128), 256, 0, stream>>>(
      ybf, w_qkv, qkv_b, nullptr, nullptr, qkvb, NTOK, D3, DIMD, 0);

  // attention -> ybf (reused as o, bf16)
  k_attn<<<NB * NH * (SEQ / 64), 128, 0, stream>>>(qkvb, ybf);

  // proj + residual(x) -> out (f32)
  k_gemm<<<dim3(DIMD / 128, NTOK / 128), 256, 0, stream>>>(
      ybf, w_prj, proj_b, x, out, nullptr, NTOK, DIMD, DIMD, 0);

  // LN2(out) -> ybf
  k_layernorm<<<NTOK / 8, 256, 0, stream>>>(out, ln2_s, ln2_b, ybf, NTOK);

  // fc1 + GELU -> hbf (bf16)
  k_gemm<<<dim3(HID / 128, NTOK / 128), 256, 0, stream>>>(
      ybf, w_fc1, fc1_b, nullptr, nullptr, hbf, NTOK, HID, DIMD, 1);

  // fc2 + residual(out) -> out (f32, element-wise read-then-write by same thread)
  k_gemm<<<dim3(DIMD / 128, NTOK / 128), 256, 0, stream>>>(
      hbf, w_fc2, fc2_b, out, out, nullptr, NTOK, DIMD, HID, 0);
}
